// SwinTransformerBlock_14319420965433
// MI455X (gfx1250) — matrix operand order repaired and checked by reading, not compiled, not run on hardware
//
#include <hip/hip_runtime.h>
#include <cmath>

// Swin transformer block for MI455X (gfx1250, wave32, WMMA f16 16x16x32).
// All 5 GEMMs (QKV, Q@K^T, P@V, proj, MLP1/2) run on v_wmma_f32_16x16x32_f16.
// V is kept transposed in LDS so every WMMA fragment load is 1-2 ds_load_b128.

using f16 = _Float16;
typedef __attribute__((ext_vector_type(16))) _Float16 v16h;
typedef __attribute__((ext_vector_type(8)))  _Float16 v8h;
typedef __attribute__((ext_vector_type(8)))  float    v8f;

constexpr int kB = 32, kC = 192, kH = 64, kW = 64;
constexpr int kHeads = 6, kHD = 32, kShift = 4;
constexpr int kN = 64;             // tokens per window (8x8)
constexpr int kNW = 64;            // windows per image
constexpr int kWinTot = kB * kNW;  // 2048
constexpr float kScale = 0.17677669529663687f; // 32^-0.5
constexpr float kEps = 1e-5f;

// ---- WMMA fragment helpers (A and B share contiguous k = 16*(lane>>4) + e) ----

// A fragment: 16x32 f16, row-major source, stride in halves.
// lane<16: M=lane, K = 0..15 ; lane>=16: M=lane-16, K = 16..31
__device__ __forceinline__ v16h frag_a(const f16* src, int stride, int lane) {
  const int m  = lane & 15;
  const int kb = (lane >> 4) << 4;
  const f16* p0 = src + m * stride + kb;
  const f16* p1 = p0 + 8;
  v16h a;
#pragma unroll
  for (int e = 0; e < 8; ++e) { a[e] = p0[e]; a[e + 8] = p1[e]; }
  return a;
}

// B fragment: 32x16 f16 from an N-major source bt[n][k].
// lane&15 = n ; K = (lane>>4)*16 + e  -> 32 contiguous bytes per lane.
__device__ __forceinline__ v16h frag_b_nmajor(const f16* bt, int stride, int lane) {
  const f16* p = bt + (lane & 15) * stride + ((lane >> 4) << 4);
  v16h b;
#pragma unroll
  for (int e = 0; e < 16; ++e) b[e] = p[e];
  return b;
}

__device__ __forceinline__ v8f wmma32(v16h a, v16h b, v8f c) {
  return __builtin_amdgcn_wmma_f32_16x16x32_f16(false, a, false, b, (short)0, c,
                                                false, false);
}

// ---- async staging of one 24 KB activation window into LDS ----
// Probe-discovered signature: (int4 AS1*, int4 AS3*, imm offset, imm cpol)
#if __has_builtin(__builtin_amdgcn_global_load_async_to_lds_b128)
#define USE_ASYNC_LDS 1
typedef int v4i __attribute__((vector_size(16)));
typedef __attribute__((address_space(1))) v4i g_v4i;
typedef __attribute__((address_space(3))) v4i l_v4i;
#else
#define USE_ASYNC_LDS 0
#endif

__device__ __forceinline__ void stage_window(f16* lds, const f16* gsrc, int tid) {
#if USE_ASYNC_LDS
  g_v4i* g = (g_v4i*)gsrc;
  l_v4i* l = (l_v4i*)lds;
  for (int i = tid; i < (kN * kC) / 8; i += 256)
    __builtin_amdgcn_global_load_async_to_lds_b128(g + i, l + i, 0, 0);
  asm volatile("s_wait_asynccnt 0x0" ::: "memory");
#else
  const uint4* src = (const uint4*)gsrc;
  uint4* dst = (uint4*)lds;
  for (int i = tid; i < (kN * kC) / 8; i += 256) dst[i] = src[i];
#endif
}

// ---- weight transpose + fp32->f16 cast: out[c][r] = in[r][c] ----
__global__ __launch_bounds__(256) void k_tcast(const float* __restrict__ in,
                                               f16* __restrict__ outp,
                                               int rows, int cols) {
  int i = blockIdx.x * 256 + threadIdx.x;
  if (i < rows * cols) {
    int r = i / cols, c2 = i - r * cols;
    outp[(size_t)c2 * rows + r] = (f16)in[(size_t)r * cols + c2];
  }
}

// ---- LN1 + roll(-4,-4) + window partition -> XW f16 (2048,64,192) ----
__global__ __launch_bounds__(256) void k_prep(const float* __restrict__ x,
                                              const float* __restrict__ g,
                                              const float* __restrict__ bta,
                                              f16* __restrict__ XW) {
  __shared__ float sx[kC * 65];  // padded stride 65 -> conflict-free column reads
  const int b = blockIdx.x >> 6;
  const int h = blockIdx.x & 63;
  const float* xp = x + (size_t)b * kC * (kH * kW) + h * kW;
  for (int i = threadIdx.x; i < (kC * kW) / 4; i += 256) {
    const int ch = i >> 4;
    const int w0 = (i & 15) * 4;
    const float4 v = *(const float4*)(xp + (size_t)ch * (kH * kW) + w0);
    float* d = &sx[ch * 65 + w0];
    d[0] = v.x; d[1] = v.y; d[2] = v.z; d[3] = v.w;
  }
  __syncthreads();
  // quad-per-token layernorm over C=192 (48 channels per lane)
  const int w = threadIdx.x >> 2;
  const int q = threadIdx.x & 3;
  const int c0 = q * 48;
  float s = 0.f;
  for (int ch = c0; ch < c0 + 48; ++ch) s += sx[ch * 65 + w];
  s += __shfl_xor(s, 1, 32); s += __shfl_xor(s, 2, 32);
  const float mu = s * (1.0f / kC);
  float vs = 0.f;
  for (int ch = c0; ch < c0 + 48; ++ch) { float d = sx[ch * 65 + w] - mu; vs += d * d; }
  vs += __shfl_xor(vs, 1, 32); vs += __shfl_xor(vs, 2, 32);
  const float inv = rsqrtf(vs * (1.0f / kC) + kEps);
  const int hr = (h - kShift) & (kH - 1);
  const int wr = (w - kShift) & (kW - 1);
  const int win = b * kNW + (hr >> 3) * 8 + (wr >> 3);
  const int n   = (hr & 7) * 8 + (wr & 7);
  f16* dst = XW + ((size_t)win * kN + n) * kC;
  for (int ch = c0; ch < c0 + 48; ++ch)
    dst[ch] = (f16)((sx[ch * 65 + w] - mu) * inv * g[ch] + bta[ch]);
}

__device__ __forceinline__ int region3(int v) { return (v < 56) ? 0 : (v < 60) ? 1 : 2; }

// ---- fused windowed attention: QKV, S+mask+bias, softmax, P@V ----
__global__ __launch_bounds__(256) void k_attn(const f16* __restrict__ XW,
                                              const f16* __restrict__ qkv_wt,
                                              const float* __restrict__ qkv_b,
                                              const float* __restrict__ rel_bias,
                                              f16* __restrict__ O) {
  __shared__ alignas(16) f16   sXW[kN * kC];   // 24576 B
  __shared__ alignas(16) f16   sQK[kN * 64];   //  8192 B : cols 0-31 Q, 32-63 K
  __shared__ alignas(16) f16   sVt[kHD * kN];  //  4096 B : V transposed [d][tok]
  __shared__ alignas(16) float sS[kN * kN];    // 16384 B
  __shared__ alignas(16) f16   sP[kN * kN];    //  8192 B  (total 61440 B)

  const int win  = blockIdx.x;
  const int lane = threadIdx.x & 31;
  const int wave = threadIdx.x >> 5;

  stage_window(sXW, XW + (size_t)win * kN * kC, threadIdx.x);
  __syncthreads();

  const int wi = win & 63;
  const int wh = wi >> 3, wwn = wi & 7;
  const int rt = wave >> 1;                       // this wave's row-tile
  const int rowbase = rt * 16 + ((lane >> 4) << 3);
  const int cl = lane & 15;

  for (int hd = 0; hd < kHeads; ++hd) {
    // ---- QKV_h (64x96): one A-fragment feeds 3 column tiles ----
    {
      v8f acc[3] = {};
      int wcol[3];
#pragma unroll
      for (int i = 0; i < 3; ++i) {
        const int c = ((wave & 1) * 3 + i) * 16 + cl;
        wcol[i] = (c >> 5) * kC + hd * kHD + (c & 31);
      }
#pragma unroll
      for (int k0 = 0; k0 < kC; k0 += 32) {
        v16h a = frag_a(sXW + rt * 16 * kC + k0, kC, lane);
#pragma unroll
        for (int i = 0; i < 3; ++i) {
          const f16* p = qkv_wt + (size_t)wcol[i] * kC + k0 + ((lane >> 4) << 4);
          v16h b;
#pragma unroll
          for (int e = 0; e < 16; ++e) b[e] = p[e];
          acc[i] = wmma32(a, b, acc[i]);
        }
      }
#pragma unroll
      for (int i = 0; i < 3; ++i) {
        const int ctq = (wave & 1) * 3 + i;
        const int c = ctq * 16 + cl;
        const float bias = qkv_b[wcol[i]];
        if (ctq >= 4) {                       // V -> transposed, vectorized store
          v8h pk;
#pragma unroll
          for (int r = 0; r < 8; ++r) pk[r] = (f16)(acc[i][r] + bias);
          *(v8h*)&sVt[(c - 64) * kN + rowbase] = pk;
        } else {                              // Q (pre-scaled) / K, token-major
          const float sc = (ctq < 2) ? kScale : 1.0f;
#pragma unroll
          for (int r = 0; r < 8; ++r)
            sQK[(rowbase + r) * 64 + c] = (f16)((acc[i][r] + bias) * sc);
        }
      }
    }
    __syncthreads();

    // ---- S = Q @ K^T (one WMMA/tile, HD=32) + mask + clipped rel bias ----
    {
      v16h a = frag_a(sQK + rt * 16 * 64, 64, lane);
#pragma unroll
      for (int i = 0; i < 2; ++i) {
        const int ct = (wave & 1) * 2 + i;
        v16h b = frag_b_nmajor(sQK + ct * 16 * 64 + 32, 64, lane);
        v8f acc = {};
        acc = wmma32(a, b, acc);
        const int m  = ct * 16 + cl;
        const int mr = m >> 3, mc = m & 7;
        const int regm = region3(wh * 8 + mr) * 3 + region3(wwn * 8 + mc);
#pragma unroll
        for (int r = 0; r < 8; ++r) {
          const int n  = rowbase + r;
          const int nr = n >> 3, nc = n & 7;
          const int regn = region3(wh * 8 + nr) * 3 + region3(wwn * 8 + nc);
          float v = acc[r] + ((regn != regm) ? -1e9f : 0.0f);
          const int idx = (nr - mr + 7) * 15 + (nc - mc + 7);
          float rb = rel_bias[idx * kHeads + hd];
          rb = fminf(fmaxf(rb, -5.f), 5.f);
          sS[n * kN + m] = fminf(fmaxf(v + rb, -10.f), 10.f);
        }
      }
    }
    __syncthreads();

    // ---- wave-parallel row softmax (8 rows/wave, 2 cols/lane) ----
    {
#pragma unroll
      for (int rr = 0; rr < 8; ++rr) {
        const int n = wave * 8 + rr;
        const float v0 = sS[n * kN + lane];
        const float v1 = sS[n * kN + 32 + lane];
        float mx = fmaxf(v0, v1);
#pragma unroll
        for (int off = 16; off; off >>= 1) mx = fmaxf(mx, __shfl_xor(mx, off, 32));
        const float e0 = expf(v0 - mx), e1 = expf(v1 - mx);
        float sm = e0 + e1;
#pragma unroll
        for (int off = 16; off; off >>= 1) sm += __shfl_xor(sm, off, 32);
        const float inv = 1.0f / sm;
        sP[n * kN + lane]      = (f16)(e0 * inv);
        sP[n * kN + 32 + lane] = (f16)(e1 * inv);
      }
    }
    __syncthreads();

    // ---- O_h = P @ V : V transposed -> contiguous B fragments ----
    {
      const int ct = wave & 1;
      v8f acc = {};
#pragma unroll
      for (int k0 = 0; k0 < kN; k0 += 32) {
        v16h a = frag_a(sP + rt * 16 * kN + k0, kN, lane);
        v16h b = frag_b_nmajor(sVt + ct * 16 * kN + k0, kN, lane);
        acc = wmma32(a, b, acc);
      }
      f16* dst = O + (size_t)win * kN * kC + hd * kHD + ct * 16 + cl;
#pragma unroll
      for (int r = 0; r < 8; ++r)
        dst[(size_t)(rowbase + r) * kC] = (f16)acc[r];
    }
    __syncthreads();  // before next head rewrites sQK/sVt/sS/sP
  }
}

// ---- proj GEMM + reverse window/roll + shortcut -> X1 (B,H,W,C) f32 ----
__global__ __launch_bounds__(256) void k_proj(const f16* __restrict__ O,
                                              const f16* __restrict__ proj_wt,
                                              const float* __restrict__ proj_b,
                                              const float* __restrict__ x,
                                              float* __restrict__ X1) {
  __shared__ alignas(16) f16 sO[kN * kC];
  const int win  = blockIdx.x;
  const int lane = threadIdx.x & 31;
  const int wave = threadIdx.x >> 5;
  stage_window(sO, O + (size_t)win * kN * kC, threadIdx.x);
  __syncthreads();
  const int b = win >> 6, wi = win & 63;
  const int wh = wi >> 3, wwn = wi & 7;
  const int rt = wave >> 1;
  const int rowbase = rt * 16 + ((lane >> 4) << 3);
  const int cl = lane & 15;

  v8f acc[6] = {};
#pragma unroll
  for (int k0 = 0; k0 < kC; k0 += 32) {
    v16h a = frag_a(sO + rt * 16 * kC + k0, kC, lane);
#pragma unroll
    for (int i = 0; i < 6; ++i) {
      const int ct = (wave & 1) * 6 + i;
      v16h bfr = frag_b_nmajor(proj_wt + (size_t)(ct * 16) * kC + k0, kC, lane);
      acc[i] = wmma32(a, bfr, acc[i]);
    }
  }
#pragma unroll
  for (int i = 0; i < 6; ++i) {
    const int c = ((wave & 1) * 6 + i) * 16 + cl;
    const float bias = proj_b[c];
#pragma unroll
    for (int r = 0; r < 8; ++r) {
      const int tok = rowbase + r;
      const int hr  = wh * 8 + (tok >> 3);
      const int wr  = wwn * 8 + (tok & 7);
      const int hh  = (hr + kShift) & (kH - 1);   // reverse roll
      const int ww2 = (wr + kShift) & (kW - 1);
      const float sc = x[(((size_t)b * kC + c) * kH + hh) * kW + ww2];
      X1[(((size_t)b * kH + hh) * kW + ww2) * kC + c] = sc + acc[i][r] + bias;
    }
  }
}

// ---- LN2 + MLP (192->768 gelu 768->192) + residual -> out (B,C,H,W) ----
__global__ __launch_bounds__(256) void k_mlp(const float* __restrict__ X1,
                                             const float* __restrict__ g,
                                             const float* __restrict__ bta,
                                             const f16* __restrict__ w1t,
                                             const float* __restrict__ b1,
                                             const f16* __restrict__ w2t,
                                             const float* __restrict__ b2,
                                             float* __restrict__ out) {
  __shared__ alignas(16) f16 sA[kN * kC];  // normalized activations, 64 tokens
  __shared__ alignas(16) f16 sH[kN * 64];  // gelu'd hidden block
  const int tok0 = blockIdx.x * kN;
  const int lane = threadIdx.x & 31;
  const int wave = threadIdx.x >> 5;
  {  // quad-per-token LN2
    const int row = threadIdx.x >> 2;
    const int q   = threadIdx.x & 3;
    const int c0  = q * 48;
    const float* xp = X1 + (size_t)(tok0 + row) * kC;
    float s = 0.f;
    for (int c2 = c0; c2 < c0 + 48; ++c2) s += xp[c2];
    s += __shfl_xor(s, 1, 32); s += __shfl_xor(s, 2, 32);
    const float mu = s * (1.0f / kC);
    float vs = 0.f;
    for (int c2 = c0; c2 < c0 + 48; ++c2) { float d = xp[c2] - mu; vs += d * d; }
    vs += __shfl_xor(vs, 1, 32); vs += __shfl_xor(vs, 2, 32);
    const float inv = rsqrtf(vs * (1.0f / kC) + kEps);
    for (int c2 = c0; c2 < c0 + 48; ++c2)
      sA[row * kC + c2] = (f16)((xp[c2] - mu) * inv * g[c2] + bta[c2]);
  }
  __syncthreads();

  const int rt = wave >> 1;
  const int rowbase = rt * 16 + ((lane >> 4) << 3);
  const int cl = lane & 15;

  v8f Y[6] = {};  // 48 output tiles distributed 6 per wave
  for (int jb = 0; jb < 4 * kC; jb += 64) {
    // H block = gelu(sA @ W1[:, jb:jb+64] + b1) : A-fragment feeds 2 tiles
    {
      v8f acc[2] = {};
#pragma unroll
      for (int k0 = 0; k0 < kC; k0 += 32) {
        v16h a = frag_a(sA + rt * 16 * kC + k0, kC, lane);
#pragma unroll
        for (int i = 0; i < 2; ++i) {
          const int ct = (wave & 1) * 2 + i;
          v16h bfr = frag_b_nmajor(w1t + (size_t)(jb + ct * 16) * kC + k0, kC, lane);
          acc[i] = wmma32(a, bfr, acc[i]);
        }
      }
#pragma unroll
      for (int i = 0; i < 2; ++i) {
        const int ct = (wave & 1) * 2 + i;
        const float bias = b1[jb + ct * 16 + cl];
#pragma unroll
        for (int r = 0; r < 8; ++r) {
          float v = acc[i][r] + bias;
          v = 0.5f * v * (1.0f + erff(v * 0.7071067811865475f));  // exact gelu
          sH[(rowbase + r) * 64 + ct * 16 + cl] = (f16)v;
        }
      }
    }
    __syncthreads();
    // Y += Hblk @ W2[jb:jb+64, :] : A-fragment feeds 6 tiles
#pragma unroll
    for (int k0 = 0; k0 < 64; k0 += 32) {
      v16h a = frag_a(sH + rt * 16 * 64 + k0, 64, lane);
#pragma unroll
      for (int i = 0; i < 6; ++i) {
        const int ct = (wave & 1) * 6 + i;
        v16h bfr = frag_b_nmajor(w2t + (size_t)(ct * 16) * (4 * kC) + jb + k0,
                                 4 * kC, lane);
        Y[i] = wmma32(a, bfr, Y[i]);
      }
    }
    __syncthreads();  // before next jb rewrites sH
  }

#pragma unroll
  for (int i = 0; i < 6; ++i) {
    const int c = ((wave & 1) * 6 + i) * 16 + cl;
    const float bias = b2[c];
#pragma unroll
    for (int r = 0; r < 8; ++r) {
      const size_t token = (size_t)tok0 + rowbase + r;
      const float v = Y[i][r] + bias + X1[token * kC + c];
      const size_t bidx = token >> 12;        // /4096
      const int hw = (int)(token & 4095);
      out[(bidx * kC + c) * 4096 + hw] = v;   // (B,C,H,W)
    }
  }
}

extern "C" void kernel_launch(void* const* d_in, const int* in_sizes, int n_in,
                              void* d_out, int out_size, void* d_ws, size_t ws_size,
                              hipStream_t stream) {
  (void)in_sizes; (void)n_in; (void)out_size; (void)ws_size;
  const float* x      = (const float*)d_in[0];
  const float* ln1_g  = (const float*)d_in[1];
  const float* ln1_b  = (const float*)d_in[2];
  const float* qkv_w  = (const float*)d_in[3];
  const float* qkv_b  = (const float*)d_in[4];
  const float* relb   = (const float*)d_in[5];
  const float* proj_w = (const float*)d_in[6];
  const float* proj_b = (const float*)d_in[7];
  const float* ln2_g  = (const float*)d_in[8];
  const float* ln2_b  = (const float*)d_in[9];
  const float* w1     = (const float*)d_in[10];
  const float* b1     = (const float*)d_in[11];
  const float* w2     = (const float*)d_in[12];
  const float* b2     = (const float*)d_in[13];
  float* out = (float*)d_out;

  char* ws = (char*)d_ws;
  f16* XW = (f16*)ws;  // (2048,64,192) f16; safely aliased as attention output O
  size_t off = (size_t)kWinTot * kN * kC * sizeof(f16);       // 50.3 MB
  float* X1 = (float*)(ws + off);                             // (B,H,W,C) f32
  off += (size_t)kB * kH * kW * kC * sizeof(float);           // +100.7 MB
  f16* qkv_wt  = (f16*)(ws + off);                            // transposed f16 weights
  f16* proj_wt = qkv_wt + 576 * 192;
  f16* w1t     = proj_wt + 192 * 192;
  f16* w2t     = w1t + 768 * 192;

  k_tcast<<<(192 * 576 + 255) / 256, 256, 0, stream>>>(qkv_w, qkv_wt, 192, 576);
  k_tcast<<<(192 * 192 + 255) / 256, 256, 0, stream>>>(proj_w, proj_wt, 192, 192);
  k_tcast<<<(192 * 768 + 255) / 256, 256, 0, stream>>>(w1, w1t, 192, 768);
  k_tcast<<<(768 * 192 + 255) / 256, 256, 0, stream>>>(w2, w2t, 768, 192);

  k_prep<<<kB * kH, 256, 0, stream>>>(x, ln1_g, ln1_b, XW);
  k_attn<<<kWinTot, 256, 0, stream>>>(XW, qkv_wt, qkv_b, relb, XW);
  k_proj<<<kWinTot, 256, 0, stream>>>(XW, proj_wt, proj_b, x, X1);
  k_mlp<<<(kB * kH * kW) / kN, 256, 0, stream>>>(X1, ln2_g, ln2_b, w1t, b1,
                                                 w2t, b2, out);
}
